// CustomSAGE_56796647522799
// MI455X (gfx1250) — matrix operand order repaired and checked by reading, not compiled, not run on hardware
//
#include <hip/hip_runtime.h>
#include <hip/hip_bf16.h>
#include <math.h>
#include <stdint.h>

// ---------------------------------------------------------------------------
// GraphSAGE (3 layers, H=128) + vocab projection (V=1000) + row softmax
// MI455X / gfx1250: wave32, WMMA bf16 16x16x32 (f32 accum), TDM tile staging.
// ---------------------------------------------------------------------------

typedef __bf16 bf16_t;
typedef __attribute__((ext_vector_type(16))) __bf16 v16bf;
typedef __attribute__((ext_vector_type(8)))  __bf16 v8bf;
typedef __attribute__((ext_vector_type(4)))  __bf16 v4bf;
typedef __attribute__((ext_vector_type(8)))  float  v8f;
typedef unsigned int v4u __attribute__((ext_vector_type(4)));
typedef int          v8i __attribute__((ext_vector_type(8)));
typedef int          v4i __attribute__((ext_vector_type(4)));

#define HID 128          // hidden size
#define AROW 136         // padded LDS row stride for bf16 staging (final kernel)
#define FROW 132         // padded LDS row stride for f32 TDM tiles (4-DW pad / 128 DW)

// ---- WMMA wrapper ---------------------------------------------------------
__device__ __forceinline__ v8f wmma_bf16(v16bf a, v16bf b, v8f c) {
  // D = A(16x32) * B(32x16) + C, f32 accumulate
  return __builtin_amdgcn_wmma_f32_16x16x32_bf16(false, a, false, b,
                                                 (short)0, c, false, false);
}

// A fragment (16x32 bf16) from a row-major bf16 [M][K] tile.
// lane (l&15)=row, half=l>>4: elems K kb..kb+15, kb=half*16.
__device__ __forceinline__ v16bf ldA(const bf16_t* p) {
  union { v16bf v; v8bf h[2]; } u;
  u.h[0] = *(const v8bf*)(p);
  u.h[1] = *(const v8bf*)(p + 8);
  return u.v;
}

// A fragment built from a row-major f32 [M][K] tile, scaled + cvt to bf16.
__device__ __forceinline__ v16bf ldA_f32(const float* p, float scale) {
  float4 a = *(const float4*)(p);
  float4 b = *(const float4*)(p + 4);
  float4 c = *(const float4*)(p + 8);
  float4 d = *(const float4*)(p + 12);
  v16bf r;
  r[0]  = (bf16_t)(a.x * scale); r[1]  = (bf16_t)(a.y * scale);
  r[2]  = (bf16_t)(a.z * scale); r[3]  = (bf16_t)(a.w * scale);
  r[4]  = (bf16_t)(b.x * scale); r[5]  = (bf16_t)(b.y * scale);
  r[6]  = (bf16_t)(b.z * scale); r[7]  = (bf16_t)(b.w * scale);
  r[8]  = (bf16_t)(c.x * scale); r[9]  = (bf16_t)(c.y * scale);
  r[10] = (bf16_t)(c.z * scale); r[11] = (bf16_t)(c.w * scale);
  r[12] = (bf16_t)(d.x * scale); r[13] = (bf16_t)(d.y * scale);
  r[14] = (bf16_t)(d.z * scale); r[15] = (bf16_t)(d.w * scale);
  return r;
}

// B fragment (32x16 bf16): lane (l&15)=col N, lanes 0-15 K0..15, 16-31 K16..31.
// Source stored transposed as [N][K] -> one contiguous 32B load.
__device__ __forceinline__ v16bf ldB(const bf16_t* p) {
  return *(const v16bf*)p;
}

// ---- TDM: async 2D f32 tile (tile_dim0=128 elems, tile_dim1=64 rows) ------
// Pads LDS dest by 4 DWORDs every 128 DWORDs -> 528B (132 f32) row stride,
// bank-conflict free for fragment reads. OOB rows (>= rowsRemain) read zero.
__device__ __forceinline__ void tdm_load_tile(void* ldsDst, const void* gsrc,
                                              unsigned rowsRemain) {
  uint64_t ga  = (uint64_t)(uintptr_t)gsrc;
  unsigned lds = (unsigned)(uintptr_t)ldsDst;   // low 32 bits = LDS byte offset
  v4u g0;
  g0[0] = 1u;                                   // count=1, user descriptor
  g0[1] = lds;                                  // lds_addr
  g0[2] = (unsigned)ga;                         // global_addr[31:0]
  g0[3] = (unsigned)((ga >> 32) & 0x01FFFFFFu)  // global_addr[56:32]
        | (2u << 30);                           // type=2 ("image")
  v8i g1;
  g1[0] = (int)((2u << 16)                      // data_size = 4 bytes
              | (1u << 20)                      // pad_enable
              | (6u << 22)                      // pad_interval: 128 DWORDs
              | (3u << 25));                    // pad_amount: 4 DWORDs (16B)
  g1[1] = (int)(128u << 16);                    // tensor_dim0 = 128 (lo16)
  g1[2] = (int)((rowsRemain & 0xFFFFu) << 16);  // tdim0 hi=0 | tensor_dim1 lo16
  g1[3] = (int)(((rowsRemain >> 16) & 0xFFFFu)  // tensor_dim1 hi16
              | (128u << 16));                  // tile_dim0 = 128
  g1[4] = 64;                                   // tile_dim1 = 64, tile_dim2 = 0
  g1[5] = 128;                                  // tensor_dim0_stride = 128 (lo32)
  g1[6] = 0;                                    // stride hi | tensor_dim1_stride lo
  g1[7] = 0;
  v4i gz4 = {0, 0, 0, 0};
  v8i gz8 = {0, 0, 0, 0, 0, 0, 0, 0};
  __builtin_amdgcn_tensor_load_to_lds(g0, g1, gz4, gz4, gz8, 0);
}

// ---- weight prep: f32 [K][N] -> bf16 [N][K] -------------------------------
__global__ void prep_wlr_kernel(const float* __restrict__ Wl,
                                const float* __restrict__ Wr,
                                bf16_t* __restrict__ wlT,
                                bf16_t* __restrict__ wrT) {
  int idx = blockIdx.x * 256 + threadIdx.x;
  const int PER = 3 * HID * HID;
  if (idx >= 2 * PER) return;
  const float* src = (idx < PER) ? Wl : Wr;
  bf16_t*      dst = (idx < PER) ? wlT : wrT;
  int i   = (idx < PER) ? idx : idx - PER;
  int l   = i / (HID * HID);
  int rem = i - l * HID * HID;
  int k   = rem >> 7;
  int n   = rem & 127;
  dst[l * HID * HID + n * HID + k] = (bf16_t)src[i];
}

__global__ void prep_wlast_kernel(const float* __restrict__ W,   // [128][V]
                                  bf16_t* __restrict__ wT,       // [1024][128]
                                  int V) {
  int idx = blockIdx.x * 256 + threadIdx.x;
  if (idx >= 1024 * HID) return;
  int n = idx >> 7, k = idx & 127;
  wT[idx] = (n < V) ? (bf16_t)W[k * V + n] : (bf16_t)0.0f;
}

// ---- embedding gather -----------------------------------------------------
__global__ void embed_kernel(const int* __restrict__ x,
                             const float* __restrict__ emb,
                             float* __restrict__ h, int N) {
  long long t = (long long)blockIdx.x * 256 + threadIdx.x;
  int n = (int)(t >> 5);
  if (n >= N) return;
  int c = ((int)t & 31) * 4;
  int tok = x[n];
  *(float4*)&h[(size_t)n * HID + c] =
      *(const float4*)&emb[(size_t)tok * HID + c];
}

// ---- per-dst degree (graph-constant) --------------------------------------
__global__ void count_kernel(const int* __restrict__ dst,
                             float* __restrict__ cnt, int E) {
  int e = blockIdx.x * 256 + threadIdx.x;
  if (e < E) atomicAdd(&cnt[dst[e]], 1.0f);
}
__global__ void inv_kernel(float* __restrict__ cnt, int N) {
  int n = blockIdx.x * 256 + threadIdx.x;
  if (n < N) cnt[n] = 1.0f / fmaxf(cnt[n], 1.0f);
}

// ---- edge scatter: agg[dst] += h[src] -------------------------------------
__global__ void scatter_kernel(const int* __restrict__ src,
                               const int* __restrict__ dst,
                               const float* __restrict__ h,
                               float* __restrict__ agg, int E) {
  long long t = (long long)blockIdx.x * 256 + threadIdx.x;
  int e = (int)(t >> 5);
  if (e >= E) return;
  int c = ((int)t & 31) * 4;
  int s = src[e], d = dst[e];
  float4 v = *(const float4*)&h[(size_t)s * HID + c];
  float* p = &agg[(size_t)d * HID + c];
  atomicAdd(p + 0, v.x);
  atomicAdd(p + 1, v.y);
  atomicAdd(p + 2, v.z);
  atomicAdd(p + 3, v.w);
}

// ---- SAGE layer GEMM: hout = relu(mean @ Wl + bl + h @ Wr) ----------------
// Block: 64 nodes x 128 cols, 256 threads (8 waves). Wave w -> cols 16w..16w+15.
// A tiles arrive via TDM as padded f32; fragments scale+convert on the fly.
__global__ __launch_bounds__(256)
void sage_gemm_kernel(const float* __restrict__ agg,
                      const float* __restrict__ hroot,
                      const float* __restrict__ inv,
                      const bf16_t* __restrict__ wlT,   // [128][128] as [N][K]
                      const bf16_t* __restrict__ wrT,
                      const float* __restrict__ bl,     // [128]
                      float* __restrict__ hout, int N) {
  __shared__ float sM32[64][FROW];   // neighbor-sum rows (f32, TDM-padded)
  __shared__ float sR32[64][FROW];   // root rows       (f32, TDM-padded)

  const int nodeBase = blockIdx.x * 64;
  const int tid  = threadIdx.x;
  const int wave = tid >> 5, lane = tid & 31;
  const int hf = lane >> 4, l16 = lane & 15;

  if (wave == 0) {
    unsigned rows = (unsigned)(N - nodeBase);   // OOB rows zero-filled by TDM
    tdm_load_tile(&sM32[0][0], agg   + (size_t)nodeBase * HID, rows);
    tdm_load_tile(&sR32[0][0], hroot + (size_t)nodeBase * HID, rows);
    __builtin_amdgcn_s_wait_tensorcnt(0);
  }
  __syncthreads();

  // per-row-tile 1/deg for the mean (lane's fragment row is rt*16 + l16)
  float invv[4];
#pragma unroll
  for (int rt = 0; rt < 4; ++rt) {
    int node = nodeBase + rt * 16 + l16;
    invv[rt] = (node < N) ? inv[node] : 0.f;
  }

  const int colBase = wave * 16;
  v8f acc[4] = {};
#pragma unroll
  for (int kc = 0; kc < 4; ++kc) {
    const int kbA = kc * 32 + hf * 16;   // A fragment K base (elements)
    const int kbB = kc * 32 + hf * 16;   // B fragment K base
    v16bf bL = ldB(&wlT[(colBase + l16) * HID + kbB]);
    v16bf bR = ldB(&wrT[(colBase + l16) * HID + kbB]);
#pragma unroll
    for (int rt = 0; rt < 4; ++rt) {
      v16bf aM = ldA_f32(&sM32[rt * 16 + l16][kbA], invv[rt]);
      v16bf aR = ldA_f32(&sR32[rt * 16 + l16][kbA], 1.0f);
      acc[rt] = wmma_bf16(aM, bL, acc[rt]);
      acc[rt] = wmma_bf16(aR, bR, acc[rt]);
    }
  }

  const float bias = bl[colBase + l16];
#pragma unroll
  for (int rt = 0; rt < 4; ++rt) {
#pragma unroll
    for (int i = 0; i < 8; ++i) {
      int node = nodeBase + rt * 16 + i + 8 * hf;
      if (node < N)
        hout[(size_t)node * HID + colBase + l16] = fmaxf(acc[rt][i] + bias, 0.f);
    }
  }
}

// ---- final: softmax(h @ Wlast + blast), fused -----------------------------
// Block: 16 nodes x 1024 (padded) cols, 256 threads (8 waves).
__global__ __launch_bounds__(256)
void final_softmax_kernel(const float* __restrict__ h,
                          const bf16_t* __restrict__ wT,   // [1024][128]
                          const float* __restrict__ blast, // [V]
                          float* __restrict__ out, int N, int V) {
  __shared__ bf16_t sH[16][AROW];
  __shared__ float  red[8][16];

  const int nodeBase = blockIdx.x * 16;
  const int tid = threadIdx.x;

#pragma unroll
  for (int it = 0; it < 2; ++it) {
    int idx = it * 256 + tid;
    int r  = idx >> 5;
    int c4 = (idx & 31) * 4;
    int node = nodeBase + r;
    float4 v = {0.f, 0.f, 0.f, 0.f};
    if (node < N) v = *(const float4*)&h[(size_t)node * HID + c4];
    v4bf p;
    p[0] = (bf16_t)v.x; p[1] = (bf16_t)v.y;
    p[2] = (bf16_t)v.z; p[3] = (bf16_t)v.w;
    *(v4bf*)&sH[r][c4] = p;
  }
  __syncthreads();

  const int wave = tid >> 5, lane = tid & 31;
  const int hf = lane >> 4, l16 = lane & 15;

  v16bf afrag[4];
#pragma unroll
  for (int kc = 0; kc < 4; ++kc)
    afrag[kc] = ldA(&sH[l16][kc * 32 + hf * 16]);

  v8f acc[8] = {};
#pragma unroll
  for (int j = 0; j < 8; ++j) {
    const int col = (wave + j * 8) * 16 + l16;
#pragma unroll
    for (int kc = 0; kc < 4; ++kc) {
      v16bf b = ldB(&wT[col * HID + kc * 32 + hf * 16]);
      acc[j] = wmma_bf16(afrag[kc], b, acc[j]);
    }
  }

  float vmax[8];
#pragma unroll
  for (int i = 0; i < 8; ++i) vmax[i] = -INFINITY;
#pragma unroll
  for (int j = 0; j < 8; ++j) {
    const int col = (wave + j * 8) * 16 + l16;
    const bool valid = (col < V);
    const float bias = valid ? blast[col] : 0.f;
#pragma unroll
    for (int i = 0; i < 8; ++i) {
      float v = valid ? (acc[j][i] + bias) : -INFINITY;
      acc[j][i] = v;
      vmax[i] = fmaxf(vmax[i], v);
    }
  }
#pragma unroll
  for (int m = 1; m <= 8; m <<= 1)
#pragma unroll
    for (int i = 0; i < 8; ++i)
      vmax[i] = fmaxf(vmax[i], __shfl_xor(vmax[i], m, 32));
  if (l16 == 0)
#pragma unroll
    for (int i = 0; i < 8; ++i) red[wave][hf * 8 + i] = vmax[i];
  __syncthreads();

  float M[8];
#pragma unroll
  for (int i = 0; i < 8; ++i) {
    float m = -INFINITY;
#pragma unroll
    for (int wv = 0; wv < 8; ++wv) m = fmaxf(m, red[wv][hf * 8 + i]);
    M[i] = m;
  }
  __syncthreads();

  float s[8];
#pragma unroll
  for (int i = 0; i < 8; ++i) s[i] = 0.f;
#pragma unroll
  for (int j = 0; j < 8; ++j)
#pragma unroll
    for (int i = 0; i < 8; ++i) {
      float e = __expf(acc[j][i] - M[i]);
      acc[j][i] = e;
      s[i] += e;
    }
#pragma unroll
  for (int m = 1; m <= 8; m <<= 1)
#pragma unroll
    for (int i = 0; i < 8; ++i) s[i] += __shfl_xor(s[i], m, 32);
  if (l16 == 0)
#pragma unroll
    for (int i = 0; i < 8; ++i) red[wave][hf * 8 + i] = s[i];
  __syncthreads();

  float S[8];
#pragma unroll
  for (int i = 0; i < 8; ++i) {
    float t = 0.f;
#pragma unroll
    for (int wv = 0; wv < 8; ++wv) t += red[wv][hf * 8 + i];
    S[i] = 1.0f / t;
  }

#pragma unroll
  for (int j = 0; j < 8; ++j) {
    const int col = (wave + j * 8) * 16 + l16;
    if (col >= V) continue;
#pragma unroll
    for (int i = 0; i < 8; ++i) {
      int node = nodeBase + i + 8 * hf;
      if (node < N) out[(size_t)node * V + col] = acc[j][i] * S[i];
    }
  }
}

// ---------------------------------------------------------------------------
extern "C" void kernel_launch(void* const* d_in, const int* in_sizes, int n_in,
                              void* d_out, int out_size, void* d_ws, size_t ws_size,
                              hipStream_t stream) {
  const int*   x     = (const int*)d_in[0];
  const int*   ei    = (const int*)d_in[1];       // [2, E]
  const float* emb   = (const float*)d_in[2];     // [V, 128]
  const float* Wl    = (const float*)d_in[3];     // [3, 128, 128]
  const float* bl    = (const float*)d_in[4];     // [3, 128]
  const float* Wr    = (const float*)d_in[5];     // [3, 128, 128]
  const float* Wlast = (const float*)d_in[6];     // [128, V]
  const float* blast = (const float*)d_in[7];     // [V]
  float* out = (float*)d_out;

  const int    N = in_sizes[0];
  const size_t E = (size_t)in_sizes[1] / 2;
  const int    V = in_sizes[7];

  // workspace layout
  float* H0  = (float*)d_ws;                      // N*128
  float* H1  = H0 + (size_t)N * HID;              // N*128 (agg, ping-pong)
  float* cnt = H1 + (size_t)N * HID;              // N (degree -> inv degree)
  size_t cntPad = ((size_t)N + 3) & ~(size_t)3;
  bf16_t* wlT  = (bf16_t*)(cnt + cntPad);         // 3*128*128
  bf16_t* wrT  = wlT + 3 * HID * HID;             // 3*128*128
  bf16_t* wlaT = wrT + 3 * HID * HID;             // 1024*128

  const int* src = ei;
  const int* dst = ei + E;

  // one-time per call: degree counts + bf16 transposed weights
  (void)hipMemsetAsync(cnt, 0, (size_t)N * sizeof(float), stream);
  prep_wlr_kernel<<<(2 * 3 * HID * HID + 255) / 256, 256, 0, stream>>>(Wl, Wr, wlT, wrT);
  prep_wlast_kernel<<<(1024 * HID + 255) / 256, 256, 0, stream>>>(Wlast, wlaT, V);
  count_kernel<<<(unsigned)((E + 255) / 256), 256, 0, stream>>>(dst, cnt, (int)E);
  inv_kernel<<<(N + 255) / 256, 256, 0, stream>>>(cnt, N);

  // h0 = emb[x]
  embed_kernel<<<(unsigned)(((size_t)N * 32 + 255) / 256), 256, 0, stream>>>(x, emb, H0, N);

  // 3 SAGE layers: scatter(H0 -> H1), gemm(H1 agg, H0 root) -> H0
  for (int l = 0; l < 3; ++l) {
    (void)hipMemsetAsync(H1, 0, (size_t)N * HID * sizeof(float), stream);
    scatter_kernel<<<(unsigned)((E * 32 + 255) / 256), 256, 0, stream>>>(src, dst, H0, H1, (int)E);
    sage_gemm_kernel<<<(N + 63) / 64, 256, 0, stream>>>(
        H1, H0, cnt, wlT + l * HID * HID, wrT + l * HID * HID, bl + l * HID, H0, N);
  }

  // projection + fused softmax
  final_softmax_kernel<<<(N + 15) / 16, 256, 0, stream>>>(H0, wlaT, blast, out, N, V);
}
